// HiveMindGNN_81836306858384
// MI455X (gfx1250) — matrix operand order repaired and checked by reading, not compiled, not run on hardware
//
#include <hip/hip_runtime.h>

#define N_NODES 100000
#define N_EDGES 1600000
#define HDIM 128
#define LN_EPS 1e-5f

typedef __bf16 bf16_t;
typedef __attribute__((ext_vector_type(16))) __bf16 v16bf;
typedef __attribute__((ext_vector_type(8)))  __bf16 v8bf;
typedef __attribute__((ext_vector_type(8)))  float  v8f;

// Force all 8 B fragments to be materialized in distinct VGPR tuples before
// the WMMA block: turns 8x (ds_load pair + s_wait_dscnt 0 + wmma) into
// 16x ds_load clause + 1 wait + 8 back-to-back wmma.
#define B_FRAG_FENCE(b) asm volatile("" \
    : "+v"(b[0]), "+v"(b[1]), "+v"(b[2]), "+v"(b[3]), \
      "+v"(b[4]), "+v"(b[5]), "+v"(b[6]), "+v"(b[7]))

static __device__ __forceinline__ float wave_reduce_sum32(float v) {
    v += __shfl_xor(v, 1, 32);
    v += __shfl_xor(v, 2, 32);
    v += __shfl_xor(v, 4, 32);
    v += __shfl_xor(v, 8, 32);
    v += __shfl_xor(v, 16, 32);
    return v;
}

// ---------------------------------------------------------------------------
// Async global->LDS 16B copy (GLOBAL_LOAD_ASYNC_TO_LDS_B128, ASYNCcnt-tracked)
// ---------------------------------------------------------------------------
typedef __attribute__((address_space(3))) char lds_char_t;

static __device__ __forceinline__ void async_copy_b128(const void* g, void* lds_dst) {
    unsigned loff = (unsigned)(unsigned long long)(lds_char_t*)lds_dst;
    asm volatile("global_load_async_to_lds_b128 %0, %1, off"
                 :: "v"(loff), "v"(g) : "memory");
}

static __device__ __forceinline__ void wait_async_then_barrier() {
    asm volatile("s_wait_asynccnt 0" ::: "memory");
    __syncthreads();
}

// ---------------------------------------------------------------------------
// One-time weight prep: emit conv_w / mlp_w1 as bf16 in *WMMA fragment order*:
// element e = ((c*8 + j)*32 + lane)*16 + i  <->  B[k = 32c + 8*(lane>>4) + (i&7) + 16*(i>>3)]
//                                               [n = 16j + (lane&15)]
// so per-block staging is a straight linear copy (async-LDS friendly) and
// B-fragment reads are fully sequential ds_load_b128 (conflict-free).
// ---------------------------------------------------------------------------
__global__ void k_prep_weights(const float* __restrict__ conv_w,
                               const float* __restrict__ mlp_w1,
                               bf16_t* __restrict__ convF,
                               bf16_t* __restrict__ w1F) {
    int idx = blockIdx.x * blockDim.x + threadIdx.x;
    if (idx < 3 * 128 * 128) {                 // conv layers: K=128 (c = 0..3)
        int l = idx >> 14;
        int e = idx & 16383;
        int i  = e & 15;
        int fl = e >> 4;
        int ln = fl & 31;
        int j  = (fl >> 5) & 7;
        int c  = fl >> 8;
        int n  = j * 16 + (ln & 15);
        int k  = c * 32 + (ln >> 4) * 8 + (i & 7) + (i >> 3) * 16;
        convF[idx] = (bf16_t)conv_w[(l << 14) + k * 128 + n];
    }
    int idx2 = idx - 3 * 128 * 128;
    if (idx2 >= 0 && idx2 < 128 * 256) {       // mlp_w1: K=256 (c = 0..7)
        int e = idx2;
        int i  = e & 15;
        int fl = e >> 4;
        int ln = fl & 31;
        int j  = (fl >> 5) & 7;
        int c  = fl >> 8;
        int n  = j * 16 + (ln & 15);
        int k  = c * 32 + (ln >> 4) * 8 + (i & 7) + (i >> 3) * 16;
        w1F[e] = (bf16_t)mlp_w1[k * 128 + n];
    }
}

// ---------------------------------------------------------------------------
// Degree (with self loop) -> dinv = rsqrt(deg)
// ---------------------------------------------------------------------------
__global__ void k_deg_init(float* __restrict__ deg) {
    int i = blockIdx.x * blockDim.x + threadIdx.x;
    if (i < N_NODES) deg[i] = 1.0f;   // self loop
}
__global__ void k_deg_count(const int* __restrict__ dst, float* __restrict__ deg) {
    int e = blockIdx.x * blockDim.x + threadIdx.x;
    if (e < N_EDGES) unsafeAtomicAdd(&deg[dst[e]], 1.0f);
}
__global__ void k_deg_rsqrt(float* __restrict__ deg) {
    int i = blockIdx.x * blockDim.x + threadIdx.x;
    if (i < N_NODES) deg[i] = rsqrtf(deg[i]);
}

// ---------------------------------------------------------------------------
// Node encoder: x = relu(LN(nf @ enc_w + enc_b)); one wave per node (wave32).
// Also mirrors x into bf16 for the WMMA GEMMs.
// ---------------------------------------------------------------------------
__global__ __launch_bounds__(256)
void k_encoder(const float* __restrict__ nf, const float* __restrict__ w,
               const float* __restrict__ b, const float* __restrict__ g,
               const float* __restrict__ bb,
               float* __restrict__ x, bf16_t* __restrict__ xb) {
    int node = (blockIdx.x * blockDim.x + threadIdx.x) >> 5;
    int lane = threadIdx.x & 31;
    if (node >= N_NODES) return;
    float fi[7];
#pragma unroll
    for (int i = 0; i < 7; i++) fi[i] = nf[node * 7 + i];
    int c0 = lane * 4;
    float v[4];
#pragma unroll
    for (int u = 0; u < 4; u++) {
        int c = c0 + u;
        float acc = b[c];
#pragma unroll
        for (int i = 0; i < 7; i++) acc += fi[i] * w[i * 128 + c];
        v[u] = acc;
    }
    float s  = wave_reduce_sum32(v[0] + v[1] + v[2] + v[3]);
    float sq = wave_reduce_sum32(v[0]*v[0] + v[1]*v[1] + v[2]*v[2] + v[3]*v[3]);
    float mu  = s * (1.0f / 128.0f);
    float var = sq * (1.0f / 128.0f) - mu * mu;
    float rs  = rsqrtf(var + LN_EPS);
    long base = (long)node * 128 + c0;
#pragma unroll
    for (int u = 0; u < 4; u++) {
        float y = fmaxf((v[u] - mu) * rs * g[c0 + u] + bb[c0 + u], 0.0f);
        x[base + u]  = y;
        xb[base + u] = (bf16_t)y;
    }
}

// ---------------------------------------------------------------------------
// h = x_bf16 @ W  (N x 128 @ 128 x 128), one 16-row tile per wave.
// W staged into LDS by async copy; per chunk all 8 B fragments preloaded into
// VGPRs (enforced by fence) so WMMAs issue back-to-back into the XDL pipe.
// ---------------------------------------------------------------------------
__global__ __launch_bounds__(256)
void k_gemm_nodes(const bf16_t* __restrict__ xb, const bf16_t* __restrict__ wF,
                  float* __restrict__ h) {
    __shared__ __align__(32) bf16_t sW[4 * 8 * 32 * 16];  // 32 KB fragment-ordered
    int tid = threadIdx.x;
    int lane = tid & 31;
    // async stage: 32KB = 2048 x 16B chunks, 8 per thread
#pragma unroll
    for (int t = 0; t < 8; t++) {
        int chunk = tid + t * 256;
        async_copy_b128(wF + chunk * 8, &sW[chunk * 8]);
    }
    wait_async_then_barrier();

    int tile = blockIdx.x * 8 + (tid >> 5);
    if (tile >= N_NODES / 16) return;
    int m = lane & 15, half = lane >> 4;
    const bf16_t* arow = xb + ((long)tile * 16 + m) * 128;
    v8f z;
#pragma unroll
    for (int i = 0; i < 8; i++) z[i] = 0.0f;
    v8f acc[8];
#pragma unroll
    for (int j = 0; j < 8; j++) acc[j] = z;
#pragma unroll
    for (int c = 0; c < 4; c++) {
        // A layout: elems 0..7 <-> k = 32c + 8*half + i ; elems 8..15 <-> +16
        v8bf a0 = *(const v8bf*)(arow + c * 32 + half * 8);
        v8bf a1 = *(const v8bf*)(arow + c * 32 + 16 + half * 8);
        v16bf af = __builtin_shufflevector(a0, a1,
            0,1,2,3,4,5,6,7,8,9,10,11,12,13,14,15);
        v16bf bf[8];
#pragma unroll
        for (int j = 0; j < 8; j++)
            bf[j] = *(const v16bf*)&sW[((c * 8 + j) * 32 + lane) * 16];
        B_FRAG_FENCE(bf);
#pragma unroll
        for (int j = 0; j < 8; j++)
            acc[j] = __builtin_amdgcn_wmma_f32_16x16x32_bf16(
                false, af, false, bf[j], (short)0, acc[j], false, false);
    }
    long rbase = (long)tile * 16 + half * 8;   // D: vgpr r <-> row m = 8*half + r
    int n = lane & 15;
#pragma unroll
    for (int j = 0; j < 8; j++)
#pragma unroll
        for (int r = 0; r < 8; r++)
            h[(rbase + r) * 128 + j * 16 + n] = acc[j][r];
}

// ---------------------------------------------------------------------------
// Edge scatter: agg[dst] += h[src] * dinv[src]*dinv[dst]   (f32 L2 atomics)
// One wave per edge; each lane handles 4 channels.
// ---------------------------------------------------------------------------
__global__ __launch_bounds__(256)
void k_scatter(const int* __restrict__ src, const int* __restrict__ dst,
               const float* __restrict__ dinv, const float* __restrict__ h,
               float* __restrict__ agg) {
    int e = (blockIdx.x * blockDim.x + threadIdx.x) >> 5;
    int lane = threadIdx.x & 31;
    if (e >= N_EDGES) return;
    int s = src[e], d = dst[e];
    float w = dinv[s] * dinv[d];
    float4 hv = *((const float4*)(h + (long)s * 128) + lane);
    float* ap = agg + (long)d * 128 + lane * 4;
    unsafeAtomicAdd(ap + 0, hv.x * w);
    unsafeAtomicAdd(ap + 1, hv.y * w);
    unsafeAtomicAdd(ap + 2, hv.z * w);
    unsafeAtomicAdd(ap + 3, hv.w * w);
}

// ---------------------------------------------------------------------------
// Post: x = relu(LN(agg + h*dinv^2 + conv_b)) + x ; refresh bf16 mirror.
// One wave per node.
// ---------------------------------------------------------------------------
__global__ __launch_bounds__(256)
void k_post(const float* __restrict__ agg, const float* __restrict__ h,
            const float* __restrict__ dinv, const float* __restrict__ cb,
            const float* __restrict__ g, const float* __restrict__ bb,
            float* __restrict__ x, bf16_t* __restrict__ xb) {
    int node = (blockIdx.x * blockDim.x + threadIdx.x) >> 5;
    int lane = threadIdx.x & 31;
    if (node >= N_NODES) return;
    float di = dinv[node];
    float sw = di * di;                      // self-loop norm
    int c0 = lane * 4;
    long base = (long)node * 128 + c0;
    float4 av = *(const float4*)(agg + base);
    float4 hv = *(const float4*)(h + base);
    float v[4] = { av.x + hv.x * sw + cb[c0 + 0],
                   av.y + hv.y * sw + cb[c0 + 1],
                   av.z + hv.z * sw + cb[c0 + 2],
                   av.w + hv.w * sw + cb[c0 + 3] };
    float s  = wave_reduce_sum32(v[0] + v[1] + v[2] + v[3]);
    float sq = wave_reduce_sum32(v[0]*v[0] + v[1]*v[1] + v[2]*v[2] + v[3]*v[3]);
    float mu  = s * (1.0f / 128.0f);
    float var = sq * (1.0f / 128.0f) - mu * mu;
    float rs  = rsqrtf(var + LN_EPS);
    float4 xo = *(const float4*)(x + base);
    float xov[4] = { xo.x, xo.y, xo.z, xo.w };
#pragma unroll
    for (int u = 0; u < 4; u++) {
        float y = fmaxf((v[u] - mu) * rs * g[c0 + u] + bb[c0 + u], 0.0f) + xov[u];
        x[base + u]  = y;
        xb[base + u] = (bf16_t)y;
    }
}

// ---------------------------------------------------------------------------
// Edge MLP: logits = relu([x[src],x[dst]] @ W1 + b1) @ w2 + b2
// 16 edges per wave; K=256 in 8 WMMA chunks; W1 fragments async-staged into
// 64KB LDS; layer-2 (H->1) fused into the epilogue with xor-shuffle reductions.
// ---------------------------------------------------------------------------
__global__ __launch_bounds__(256)
void k_edge_mlp(const int* __restrict__ src, const int* __restrict__ dst,
                const bf16_t* __restrict__ xb, const bf16_t* __restrict__ w1F,
                const float* __restrict__ b1, const float* __restrict__ w2,
                const float* __restrict__ b2, float* __restrict__ out) {
    __shared__ __align__(32) bf16_t sW[8 * 8 * 32 * 16];  // 64 KB fragment-ordered
    int tid = threadIdx.x;
    int lane = tid & 31;
    // async stage: 64KB = 4096 x 16B chunks, 16 per thread
#pragma unroll
    for (int t = 0; t < 16; t++) {
        int chunk = tid + t * 256;
        async_copy_b128(w1F + chunk * 8, &sW[chunk * 8]);
    }
    wait_async_then_barrier();

    int tile = blockIdx.x * 8 + (tid >> 5);
    if (tile >= N_EDGES / 16) return;
    int m = lane & 15, half = lane >> 4;
    int e = tile * 16 + m;
    long si = src[e], di = dst[e];
    v8f z;
#pragma unroll
    for (int i = 0; i < 8; i++) z[i] = 0.0f;
    v8f acc[8];
#pragma unroll
    for (int j = 0; j < 8; j++) acc[j] = z;
#pragma unroll
    for (int c = 0; c < 8; c++) {
        const bf16_t* arow = (c < 4) ? (xb + si * 128 + c * 32)
                                     : (xb + di * 128 + (c - 4) * 32);
        v8bf a0 = *(const v8bf*)(arow + half * 8);
        v8bf a1 = *(const v8bf*)(arow + 16 + half * 8);
        v16bf af = __builtin_shufflevector(a0, a1,
            0,1,2,3,4,5,6,7,8,9,10,11,12,13,14,15);
        v16bf bf[8];
#pragma unroll
        for (int j = 0; j < 8; j++)
            bf[j] = *(const v16bf*)&sW[((c * 8 + j) * 32 + lane) * 16];
        B_FRAG_FENCE(bf);
#pragma unroll
        for (int j = 0; j < 8; j++)
            acc[j] = __builtin_amdgcn_wmma_f32_16x16x32_bf16(
                false, af, false, bf[j], (short)0, acc[j], false, false);
    }
    int n = lane & 15;
    float b1l[8], w2l[8];
#pragma unroll
    for (int j = 0; j < 8; j++) { b1l[j] = b1[j * 16 + n]; w2l[j] = w2[j * 16 + n]; }
    float bias2 = b2[0];
#pragma unroll
    for (int r = 0; r < 8; r++) {
        float p = 0.0f;
#pragma unroll
        for (int j = 0; j < 8; j++) {
            float y = fmaxf(acc[j][r] + b1l[j], 0.0f);
            p += y * w2l[j];
        }
        // reduce over the 16 lanes of each half (rows m=r and m=8+r)
        p += __shfl_xor(p, 1, 32);
        p += __shfl_xor(p, 2, 32);
        p += __shfl_xor(p, 4, 32);
        p += __shfl_xor(p, 8, 32);
        if (lane == 0)  out[tile * 16 + r]     = p + bias2;
        if (lane == 16) out[tile * 16 + 8 + r] = p + bias2;
    }
}

// ---------------------------------------------------------------------------
extern "C" void kernel_launch(void* const* d_in, const int* in_sizes, int n_in,
                              void* d_out, int out_size, void* d_ws, size_t ws_size,
                              hipStream_t stream) {
    const float* nf     = (const float*)d_in[0];
    const int*   eidx   = (const int*)d_in[1];
    // d_in[2] = edge_attr (unused by the reference)
    const float* enc_w  = (const float*)d_in[3];
    const float* enc_b  = (const float*)d_in[4];
    const float* enc_g  = (const float*)d_in[5];
    const float* enc_bb = (const float*)d_in[6];
    const float* conv_w = (const float*)d_in[7];
    const float* conv_b = (const float*)d_in[8];
    const float* ln_g   = (const float*)d_in[9];
    const float* ln_b   = (const float*)d_in[10];
    const float* w1     = (const float*)d_in[11];
    const float* b1     = (const float*)d_in[12];
    const float* w2     = (const float*)d_in[13];
    const float* b2     = (const float*)d_in[14];

    const int* srcI = eidx;
    const int* dstI = eidx + N_EDGES;

    float* x      = (float*)d_out;
    float* logits = (float*)d_out + (long)N_NODES * HDIM;

    char* p = (char*)d_ws;
    auto take = [&](size_t bytes) {
        char* q = p;
        p += (bytes + 255) & ~(size_t)255;
        return q;
    };
    float*  h     = (float*)take((size_t)N_NODES * 128 * 4);
    float*  agg   = (float*)take((size_t)N_NODES * 128 * 4);
    bf16_t* xbm   = (bf16_t*)take((size_t)N_NODES * 128 * 2);
    float*  dinv  = (float*)take((size_t)N_NODES * 4);
    bf16_t* convF = (bf16_t*)take((size_t)3 * 128 * 128 * 2);
    bf16_t* w1F   = (bf16_t*)take((size_t)128 * 256 * 2);

    k_prep_weights<<<320, 256, 0, stream>>>(conv_w, w1, convF, w1F);
    k_deg_init<<<(N_NODES + 255) / 256, 256, 0, stream>>>(dinv);
    k_deg_count<<<(N_EDGES + 255) / 256, 256, 0, stream>>>(dstI, dinv);
    k_deg_rsqrt<<<(N_NODES + 255) / 256, 256, 0, stream>>>(dinv);
    k_encoder<<<N_NODES / 8, 256, 0, stream>>>(nf, enc_w, enc_b, enc_g, enc_bb, x, xbm);

    for (int l = 0; l < 3; l++) {
        hipMemsetAsync(agg, 0, (size_t)N_NODES * 128 * 4, stream);
        k_gemm_nodes<<<(N_NODES / 16 + 7) / 8, 256, 0, stream>>>(
            xbm, convF + (size_t)l * 128 * 128, h);
        k_scatter<<<N_EDGES / 8, 256, 0, stream>>>(srcI, dstI, dinv, h, agg);
        k_post<<<N_NODES / 8, 256, 0, stream>>>(
            agg, h, dinv, conv_b + l * 128, ln_g + l * 128, ln_b + l * 128, x, xbm);
    }

    k_edge_mlp<<<N_EDGES / 16 / 8, 256, 0, stream>>>(srcI, dstI, xbm, w1F, b1, w2, b2, logits);
}
